// AdvancedVortexGNN_42949672960269
// MI455X (gfx1250) — matrix operand order repaired and checked by reading, not compiled, not run on hardware
//
#include <hip/hip_runtime.h>
#include <hip/hip_bf16.h>
#include <math.h>

// ---------------------------------------------------------------------------
// Types for CDNA5 WMMA (wave32, 16x16x32 f16 -> f32)
// ---------------------------------------------------------------------------
typedef _Float16 f16;
typedef __attribute__((ext_vector_type(16))) _Float16 v16h;
typedef __attribute__((ext_vector_type(8)))  _Float16 v8h;
typedef __attribute__((ext_vector_type(8)))  float    v8f;

// Pointer types for the CDNA5 async global->LDS builtin:
// param0: v4i in AS1 (global), param1: v4i in AS3 (LDS)
typedef __attribute__((vector_size(16))) int i32x4;
typedef __attribute__((address_space(1))) i32x4 as1_i32x4;
typedef __attribute__((address_space(3))) i32x4 as3_i32x4;

#define TM 128
#define TN 128
#define TK 32

// CDNA5 async global->LDS copy path (tracked on ASYNCcnt), guarded so the
// fallback (sync load + ds_store) is used if this toolchain lacks the builtins.
#if defined(__has_builtin)
#if __has_builtin(__builtin_amdgcn_global_load_async_to_lds_b128) && \
    __has_builtin(__builtin_amdgcn_s_wait_asynccnt)
#define USE_ASYNC_LDS 1
#else
#define USE_ASYNC_LDS 0
#endif
#else
#define USE_ASYNC_LDS 0
#endif

enum { EPI_NONE = 0, EPI_BIAS = 1, EPI_TANH = 2, EPI_LRELU = 3 };

// ---------------------------------------------------------------------------
// Tiled WMMA GEMM:  C[M,N] = epi(A[M,K] @ B[K,N] + bias)
// A: f16 row-major with leading dim lda.  Bt: f16 *transposed* (N x K).
// 256 threads = 8 waves; block tile 128x128; wave tile 64x32 (4x2 WMMA tiles).
// Double-buffered LDS, software-pipelined: stage tile k+1 while computing k.
// Assumes M%128==0, N%128==0, K%32==0 (true for all call sites here).
// ---------------------------------------------------------------------------
template <int EPI, typename OutT>
__global__ __launch_bounds__(256)
void wmma_gemm_kernel(const f16* __restrict__ A, int lda,
                      const f16* __restrict__ Bt,
                      const float* __restrict__ bias,
                      OutT* __restrict__ C, int ldc,
                      int M, int N, int K)
{
    __shared__ f16 As[2][TM][TK + 8];   // +8 halves pad against bank conflicts
    __shared__ f16 Bs[2][TN][TK + 8];

    const int tid  = threadIdx.x;
    const int lane = tid & 31;
    const int wave = tid >> 5;
    const int wm   = wave & 1;      // 2 waves along M (64 rows each)
    const int wn   = wave >> 1;     // 4 waves along N (32 cols each)

    const int bm = blockIdx.x * TM;
    const int bn = blockIdx.y * TN;

    v8f acc[4][2];
#pragma unroll
    for (int i = 0; i < 4; ++i)
#pragma unroll
        for (int j = 0; j < 2; ++j)
            acc[i][j] = (v8f){0.f, 0.f, 0.f, 0.f, 0.f, 0.f, 0.f, 0.f};

    const int arow = lane & 15;     // row (A) / col (B) within 16-wide tile
    const int ksel = lane >> 4;     // K-half select per ISA VGPR layout

    // Stage one 128x32 A tile + 128x32 Bt tile into LDS buffer `buf`.
    // chunk = c*256 + tid -> consecutive lanes move consecutive 16B (coalesced).
    auto stage = [&](int buf, int k0) {
#pragma unroll
        for (int c = 0; c < 2; ++c) {
            const int chunk = c * 256 + tid;    // 0..511
            const int r = chunk >> 2;           // 0..127
            const int o = (chunk & 3) * 8;      // half offset 0/8/16/24
            const f16* gA = A  + (size_t)(bm + r) * lda + k0 + o;
            const f16* gB = Bt + (size_t)(bn + r) * K   + k0 + o;
            f16* lA = &As[buf][r][o];
            f16* lB = &Bs[buf][r][o];
#if USE_ASYNC_LDS
            __builtin_amdgcn_global_load_async_to_lds_b128(
                (as1_i32x4*)gA, (as3_i32x4*)lA, 0, 0);
            __builtin_amdgcn_global_load_async_to_lds_b128(
                (as1_i32x4*)gB, (as3_i32x4*)lB, 0, 0);
#else
            *(v8h*)lA = *(const v8h*)gA;
            *(v8h*)lB = *(const v8h*)gB;
#endif
        }
    };

    stage(0, 0);
    int cur = 0;
    for (int k0 = 0; k0 < K; k0 += TK) {
#if USE_ASYNC_LDS
        __builtin_amdgcn_s_wait_asynccnt(0);   // own async stores to LDS landed
#endif
        __syncthreads();                        // everyone's staging visible
        // All waves passed the barrier, so all finished computing from
        // buffer cur^1 in the previous iteration -> safe to overwrite it now.
        if (k0 + TK < K) stage(cur ^ 1, k0 + TK);

        // ---- build fragments from LDS buffer `cur` (contiguous 16B reads) ----
        v16h afrag[4];
#pragma unroll
        for (int mt = 0; mt < 4; ++mt) {
            const f16* p = &As[cur][wm * 64 + mt * 16 + arow][ksel * 8];
            v8h lo = *(const v8h*)p;
            v8h hi = *(const v8h*)(p + 16);
            afrag[mt] = __builtin_shufflevector(lo, hi,
                0, 1, 2, 3, 4, 5, 6, 7, 8, 9, 10, 11, 12, 13, 14, 15);
        }
        v16h bfrag[2];
#pragma unroll
        for (int nt = 0; nt < 2; ++nt) {
            const f16* p = &Bs[cur][wn * 32 + nt * 16 + arow][ksel * 8];
            v8h lo = *(const v8h*)p;
            v8h hi = *(const v8h*)(p + 16);
            bfrag[nt] = __builtin_shufflevector(lo, hi,
                0, 1, 2, 3, 4, 5, 6, 7, 8, 9, 10, 11, 12, 13, 14, 15);
        }

        // ---- 8 WMMAs per wave per K-step ----
#pragma unroll
        for (int mt = 0; mt < 4; ++mt)
#pragma unroll
            for (int nt = 0; nt < 2; ++nt)
                acc[mt][nt] = __builtin_amdgcn_wmma_f32_16x16x32_f16(
                    false, afrag[mt], false, bfrag[nt],
                    (short)0, acc[mt][nt], false, false);

        cur ^= 1;
    }

    // ---- epilogue: C/D layout -> row = 8*(lane>=16)+r, col = lane&15 ----
#pragma unroll
    for (int mt = 0; mt < 4; ++mt) {
#pragma unroll
        for (int nt = 0; nt < 2; ++nt) {
            const int col  = bn + wn * 32 + nt * 16 + (lane & 15);
            const int row0 = bm + wm * 64 + mt * 16 + (lane >> 4) * 8;
            float bv = 0.0f;
            if (EPI != EPI_NONE) bv = bias[col];
#pragma unroll
            for (int r = 0; r < 8; ++r) {
                float val = acc[mt][nt][r] + bv;
                if (EPI == EPI_TANH)  val = tanhf(val);
                if (EPI == EPI_LRELU) val = (val > 0.0f) ? val : 0.01f * val;
                C[(size_t)(row0 + r) * ldc + col] = (OutT)val;
            }
        }
    }
}

// ---------------------------------------------------------------------------
// Small helper kernels
// ---------------------------------------------------------------------------
__global__ void k_f32_to_f16(const float* __restrict__ s, f16* __restrict__ d, int n)
{
    int i = blockIdx.x * 256 + threadIdx.x;
    if (i < n) d[i] = (f16)s[i];
}

// W (K x N, f32, row-major) -> Wt (N x K, f16)
__global__ void k_cvt_T(const float* __restrict__ W, f16* __restrict__ Wt, int K, int N)
{
    int i = blockIdx.x * 256 + threadIdx.x;
    if (i >= K * N) return;
    int k = i / N, n = i % N;
    Wt[(size_t)n * K + k] = (f16)W[i];
}

// Assemble 512x512 Hamilton-product block matrix, stored transposed (N x K).
__global__ void k_quat_T(const float* __restrict__ Wr, const float* __restrict__ Wi,
                         const float* __restrict__ Wj, const float* __restrict__ Wk,
                         f16* __restrict__ Qt)
{
    int i = blockIdx.x * 256 + threadIdx.x;         // over 512*512
    if (i >= 512 * 512) return;
    int k = i >> 9, n = i & 511;                    // k = input dim, n = output dim
    int bi = k >> 7, bo = n >> 7, kk = k & 127, nn = n & 127;
    const int   whichT[16] = {0,1,2,3,  1,0,3,2,  2,3,0,1,  3,2,1,0};
    const float signT [16] = {1,1,1,1, -1,1,-1,1, -1,1,1,-1, -1,-1,1,1};
    const float* Ws[4] = {Wr, Wi, Wj, Wk};
    int s = bi * 4 + bo;
    float v = signT[s] * Ws[whichT[s]][kk * 128 + nn];
    Qt[(size_t)n * 512 + k] = (f16)v;
}

// Assemble 512x512 wavelet matrix (Haar split + per-band mix + merge),
// stored transposed (N x K).
__global__ void k_wav_T(const float* __restrict__ Wlo, const float* __restrict__ Whi,
                        f16* __restrict__ Wt)
{
    int i = blockIdx.x * 256 + threadIdx.x;
    if (i >= 512 * 512) return;
    int k = i >> 9, n = i & 511;
    int j = k >> 1, p = k & 1, m = n >> 1, q = n & 1;
    float lo = Wlo[j * 256 + m], hi = Whi[j * 256 + m];
    float v = 0.5f * ((p == q) ? (lo + hi) : (lo - hi));
    Wt[(size_t)n * 512 + k] = (f16)v;
}

// Vortex aggregation: m[b,r,:] = 0.5*(nf[b,r,:]+nf[b,src[r],:]) (r<8), m[b,8]=nf[b,8]
__global__ void k_agg(const f16* __restrict__ nf, f16* __restrict__ m, int BH)
{
    int i = blockIdx.x * 256 + threadIdx.x;         // over B*H
    if (i >= BH) return;
    int b = i >> 9, h = i & 511;
    const f16* in  = nf + (size_t)b * 9 * 512 + h;
    f16*       out = m  + (size_t)b * 9 * 512 + h;
    float x[9];
#pragma unroll
    for (int n = 0; n < 9; ++n) x[n] = (float)in[n * 512];
    const int src[8] = {4, 0, 5, 1, 6, 2, 7, 3};
#pragma unroll
    for (int r = 0; r < 8; ++r) out[r * 512] = (f16)(0.5f * (x[r] + x[src[r]]));
    out[8 * 512] = (f16)x[8];
}

// Attention core over 9 nodes: o = softmax(q k^T / sqrt(512)) v   (per batch)
__global__ __launch_bounds__(256)
void k_attn_core(const f16* __restrict__ q, const f16* __restrict__ k,
                 const f16* __restrict__ v, f16* __restrict__ o)
{
    __shared__ float S[9][9];
    const size_t base = (size_t)blockIdx.x * 9 * 512;
    const f16* qb = q + base; const f16* kb = k + base;
    const f16* vb = v + base; f16*       ob = o + base;
    const int t = threadIdx.x;

    if (t < 81) {
        int qi = t / 9, ki = t % 9;
        float s = 0.f;
        for (int h = 0; h < 512; ++h)
            s += (float)qb[qi * 512 + h] * (float)kb[ki * 512 + h];
        S[qi][ki] = s * 0.04419417382f;      // 1/sqrt(512)
    }
    __syncthreads();
    if (t < 9) {
        float mx = -1e30f;
        for (int j2 = 0; j2 < 9; ++j2) mx = fmaxf(mx, S[t][j2]);
        float sum = 0.f;
        for (int j2 = 0; j2 < 9; ++j2) { float e = expf(S[t][j2] - mx); S[t][j2] = e; sum += e; }
        float inv = 1.f / sum;
        for (int j2 = 0; j2 < 9; ++j2) S[t][j2] *= inv;
    }
    __syncthreads();
    for (int idx = t; idx < 9 * 512; idx += 256) {
        int qi = idx >> 9, h = idx & 511;
        float s = 0.f;
#pragma unroll
        for (int j2 = 0; j2 < 9; ++j2) s += S[qi][j2] * (float)vb[j2 * 512 + h];
        ob[idx] = (f16)s;
    }
}

// nf = v + gamma*(a-mu)*rsqrt(var+eps) + beta, per 512-wide row
__global__ __launch_bounds__(256)
void k_drnorm(const f16* __restrict__ a, const f16* __restrict__ v,
              const float* __restrict__ gamma, const float* __restrict__ beta,
              f16* __restrict__ out)
{
    __shared__ float s1[256], s2[256];
    const size_t row = blockIdx.x;
    const f16* ar = a + row * 512;
    const f16* vr = v + row * 512;
    f16*      orow = out + row * 512;
    const int t = threadIdx.x;
    float x0 = (float)ar[t], x1 = (float)ar[t + 256];
    s1[t] = x0 + x1;
    s2[t] = x0 * x0 + x1 * x1;
    __syncthreads();
    for (int off = 128; off > 0; off >>= 1) {
        if (t < off) { s1[t] += s1[t + off]; s2[t] += s2[t + off]; }
        __syncthreads();
    }
    float mu  = s1[0] * (1.0f / 512.0f);
    float var = s2[0] * (1.0f / 512.0f) - mu * mu;
    float rs  = rsqrtf(var + 1e-5f);
    orow[t]       = (f16)((float)vr[t]       + gamma[t]       * (x0 - mu) * rs + beta[t]);
    orow[t + 256] = (f16)((float)vr[t + 256] + gamma[t + 256] * (x1 - mu) * rs + beta[t + 256]);
}

// ---------------------------------------------------------------------------
// Host orchestration
// ---------------------------------------------------------------------------
extern "C" void kernel_launch(void* const* d_in, const int* in_sizes, int n_in,
                              void* d_out, int out_size, void* d_ws, size_t ws_size,
                              hipStream_t stream)
{
    const float* x      = (const float*)d_in[0];
    const float* W_in   = (const float*)d_in[1];
    const float* b_in   = (const float*)d_in[2];
    const float* W_dr   = (const float*)d_in[3];
    const float* b_dr   = (const float*)d_in[4];
    const float* W_hyp  = (const float*)d_in[5];
    const float* b_hyp  = (const float*)d_in[6];
    const float* Wq_r   = (const float*)d_in[7];
    const float* Wq_i   = (const float*)d_in[8];
    const float* Wq_j   = (const float*)d_in[9];
    const float* Wq_k   = (const float*)d_in[10];
    const float* b_quat = (const float*)d_in[11];
    const float* W_low  = (const float*)d_in[12];
    const float* W_high = (const float*)d_in[13];
    const float* b_wav  = (const float*)d_in[14];
    const float* aWq    = (const float*)d_in[15];
    const float* aWk    = (const float*)d_in[16];
    const float* aWv    = (const float*)d_in[17];
    const float* aWo    = (const float*)d_in[18];
    const float* gamma  = (const float*)d_in[19];
    const float* beta   = (const float*)d_in[20];
    const float* W_out  = (const float*)d_in[21];
    const float* b_out  = (const float*)d_in[22];

    const int B = 16384, D = 512, H = 512, O = 128;
    const int R = B * 9;              // 147456 GEMM rows
    const int BH = B * H;

    // ---- workspace carve ----
    size_t off = 0;
    auto carve = [&](size_t bytes) -> char* {
        char* p = (char*)d_ws + off;
        off = (off + bytes + 255) & ~(size_t)255;
        return p;
    };
    f16* xh    = (f16*)carve((size_t)B * D * 2);
    f16* wInT  = (f16*)carve((size_t)9 * H * D * 2);
    f16* wDrT  = (f16*)carve((size_t)H * H * 2);
    f16* wHypT = (f16*)carve((size_t)H * H * 2);
    f16* attnT = (f16*)carve((size_t)12 * H * H * 2);   // [l][q,k,v,o]
    f16* quatT = (f16*)carve((size_t)H * H * 2);
    f16* wavT  = (f16*)carve((size_t)H * H * 2);
    f16* wOutT = (f16*)carve((size_t)O * H * 2);
    f16* act0  = (f16*)carve((size_t)R * H * 2);
    f16* act1  = (f16*)carve((size_t)R * H * 2);
    f16* act2  = (f16*)carve((size_t)R * H * 2);
    f16* act3  = (f16*)carve((size_t)R * H * 2);
    f16* act4  = (f16*)carve((size_t)R * H * 2);

    auto blocks = [](int n) { return (n + 255) / 256; };

    // ---- weight conversion / assembly (f16, transposed for GEMM B) ----
    k_f32_to_f16<<<blocks(B * D), 256, 0, stream>>>(x, xh, B * D);
    k_cvt_T<<<blocks(D * 9 * H), 256, 0, stream>>>(W_in, wInT, D, 9 * H);
    k_cvt_T<<<blocks(H * H), 256, 0, stream>>>(W_dr,  wDrT,  H, H);
    k_cvt_T<<<blocks(H * H), 256, 0, stream>>>(W_hyp, wHypT, H, H);
    for (int l = 0; l < 3; ++l) {
        k_cvt_T<<<blocks(H * H), 256, 0, stream>>>(aWq + (size_t)l * H * H, attnT + (size_t)(l * 4 + 0) * H * H, H, H);
        k_cvt_T<<<blocks(H * H), 256, 0, stream>>>(aWk + (size_t)l * H * H, attnT + (size_t)(l * 4 + 1) * H * H, H, H);
        k_cvt_T<<<blocks(H * H), 256, 0, stream>>>(aWv + (size_t)l * H * H, attnT + (size_t)(l * 4 + 2) * H * H, H, H);
        k_cvt_T<<<blocks(H * H), 256, 0, stream>>>(aWo + (size_t)l * H * H, attnT + (size_t)(l * 4 + 3) * H * H, H, H);
    }
    k_quat_T<<<blocks(H * H), 256, 0, stream>>>(Wq_r, Wq_i, Wq_j, Wq_k, quatT);
    k_wav_T<<<blocks(H * H), 256, 0, stream>>>(W_low, W_high, wavT);
    k_cvt_T<<<blocks(H * O), 256, 0, stream>>>(W_out, wOutT, H, O);

    // ---- p = leaky_relu(x @ W_in + b_in) : (16384x512)@(512x4608) ----
    wmma_gemm_kernel<EPI_LRELU, f16><<<dim3(B / TM, (9 * H) / TN), 256, 0, stream>>>(
        xh, D, wInT, b_in, act0, 9 * H, B, 9 * H, D);

    // ---- nf = tanh(p @ W_dr + b_dr) ----
    wmma_gemm_kernel<EPI_TANH, f16><<<dim3(R / TM, H / TN), 256, 0, stream>>>(
        act0, H, wDrT, b_dr, act1, H, R, H, H);

    // ---- m = agg(nf); v = tanh(m @ W_hyp + b_hyp) ----
    k_agg<<<blocks(BH), 256, 0, stream>>>(act1, act2, BH);
    wmma_gemm_kernel<EPI_TANH, f16><<<dim3(R / TM, H / TN), 256, 0, stream>>>(
        act2, H, wHypT, b_hyp, act0, H, R, H, H);

    // ---- attention block helper (v in act0; nf out in act2) ----
    auto attention_block = [&](int l) {
        const f16* Wq_t = attnT + (size_t)(l * 4 + 0) * H * H;
        const f16* Wk_t = attnT + (size_t)(l * 4 + 1) * H * H;
        const f16* Wv_t = attnT + (size_t)(l * 4 + 2) * H * H;
        const f16* Wo_t = attnT + (size_t)(l * 4 + 3) * H * H;
        wmma_gemm_kernel<EPI_NONE, f16><<<dim3(R / TM, H / TN), 256, 0, stream>>>(
            act0, H, Wq_t, nullptr, act1, H, R, H, H);
        wmma_gemm_kernel<EPI_NONE, f16><<<dim3(R / TM, H / TN), 256, 0, stream>>>(
            act0, H, Wk_t, nullptr, act2, H, R, H, H);
        wmma_gemm_kernel<EPI_NONE, f16><<<dim3(R / TM, H / TN), 256, 0, stream>>>(
            act0, H, Wv_t, nullptr, act3, H, R, H, H);
        k_attn_core<<<B, 256, 0, stream>>>(act1, act2, act3, act4);
        wmma_gemm_kernel<EPI_NONE, f16><<<dim3(R / TM, H / TN), 256, 0, stream>>>(
            act4, H, Wo_t, nullptr, act1, H, R, H, H);
        k_drnorm<<<R, 256, 0, stream>>>(act1, act0, gamma + (size_t)l * H,
                                        beta + (size_t)l * H, act2);
    };

    // ---- layer 0 ----
    attention_block(0);

    // ---- quaternion mix: v = tanh(agg(nf) @ Q + b_quat) ----
    k_agg<<<blocks(BH), 256, 0, stream>>>(act2, act1, BH);
    wmma_gemm_kernel<EPI_TANH, f16><<<dim3(R / TM, H / TN), 256, 0, stream>>>(
        act1, H, quatT, b_quat, act0, H, R, H, H);

    // ---- layer 1 ----
    attention_block(1);

    // ---- wavelet mix: v = tanh(agg(nf) @ W_wav + b_wav) ----
    k_agg<<<blocks(BH), 256, 0, stream>>>(act2, act1, BH);
    wmma_gemm_kernel<EPI_TANH, f16><<<dim3(R / TM, H / TN), 256, 0, stream>>>(
        act1, H, wavT, b_wav, act0, H, R, H, H);

    // ---- layer 2 ----
    attention_block(2);

    // ---- out = nf[:,8,:] @ W_out + b_out : rows strided by 9*512 ----
    wmma_gemm_kernel<EPI_BIAS, float><<<dim3(B / TM, O / TN), 256, 0, stream>>>(
        act2 + (size_t)8 * H, 9 * H, wOutT, b_out, (float*)d_out, O, B, O, H);
}
